// CrossAttention_7937099563211
// MI455X (gfx1250) — matrix operand order repaired and checked by reading, not compiled, not run on hardware
//
#include <hip/hip_runtime.h>

// ---------------------------------------------------------------------------
// CrossAttention for MI455X (gfx1250, wave32, WMMA + async-to-LDS DMA).
//   pre-convert Q,K,V,Wk,Wv,Wo -> bf16 (once);
//   K_proj = K@Wk^T+bk ; V_projT = (V@Wv^T+bv)^T ; S = Q@K_proj^T/32 ;
//   P = softmax(S) (bf16, in place) ; ctx = P@V_proj ; out = ctx@Wo^T+bo
// All GEMMs: v_wmma_f32_16x16x32_bf16, 128x64x32 tiles, 8 waves/WG,
// double-buffered LDS refilled by global_load_async_to_lds_b128 (ASYNCcnt).
// ---------------------------------------------------------------------------

typedef __attribute__((ext_vector_type(16))) __bf16 v16bf;
typedef __attribute__((ext_vector_type(8)))  __bf16 v8bf;
typedef __attribute__((ext_vector_type(8)))  float  v8f;
typedef unsigned short u16_t;

static __device__ __forceinline__ u16_t f2bf_bits(float f) {
  union { float f; unsigned u; } v; v.f = f;
  // round-to-nearest-even truncation to bf16
  return (u16_t)((v.u + 0x7FFFu + ((v.u >> 16) & 1u)) >> 16);
}
static __device__ __forceinline__ unsigned pack_bf2(float a, float b) {
  return (unsigned)f2bf_bits(a) | ((unsigned)f2bf_bits(b) << 16);
}

// generic pointer -> wave-relative LDS byte address (low 32 bits of the
// LDS-aperture generic address)
static __device__ __forceinline__ unsigned lds_addr(const void* p) {
  return (unsigned)(unsigned long long)p;
}
// 32B per lane (two b128): INST_OFFSET applies to both LDS and global address.
static __device__ __forceinline__ void async_copy_b128x2(unsigned lds, const void* g) {
  asm volatile("global_load_async_to_lds_b128 %0, %1, off\n\t"
               "global_load_async_to_lds_b128 %0, %1, off offset:16"
               :: "v"(lds), "v"(g) : "memory");
}
static __device__ __forceinline__ void async_copy_b128(unsigned lds, const void* g) {
  asm volatile("global_load_async_to_lds_b128 %0, %1, off"
               :: "v"(lds), "v"(g) : "memory");
}
static __device__ __forceinline__ void wait_async0() {
  asm volatile("s_wait_asynccnt 0x0" ::: "memory");
}

// One-time f32 -> bf16 conversion, 8 elems/thread, fully coalesced.
__global__ __launch_bounds__(256) void convert_f32_bf16(
    const float* __restrict__ src, u16_t* __restrict__ dst, int n) {
  const int i = (blockIdx.x * 256 + threadIdx.x) * 8;
  if (i < n) {
    float4 a = ((const float4*)(src + i))[0];
    float4 b = ((const float4*)(src + i))[1];
    uint4 o;
    o.x = pack_bf2(a.x, a.y);
    o.y = pack_bf2(a.z, a.w);
    o.z = pack_bf2(b.x, b.y);
    o.w = pack_bf2(b.z, b.w);
    *(uint4*)(dst + i) = o;
  }
}

// OUT_MODE: 0 = f32 row-major, 1 = bf16 row-major, 2 = bf16 transposed ([N][M])
// A: [M][K] bf16 (lda), B: [N][K] bf16 (ldb) -- both K-contiguous.
template <int OUT_MODE, bool HAS_BIAS>
__global__ __launch_bounds__(256) void gemm_bf16_wmma(
    const u16_t* __restrict__ A, const u16_t* __restrict__ B,
    void* __restrict__ Cp, const float* __restrict__ bias,
    int M, int N, int K, int lda, int ldb, int ldc,
    long long sA, long long sB, long long sC, float alpha) {
  constexpr int BM = 128, BN = 64, BK = 32;
  constexpr int SA = 40, SB = 40;  // padded LDS strides (bf16 elems), 80B rows
  __shared__ u16_t AsU[2][BM * SA];
  __shared__ u16_t BsU[2][BN * SB];

  const int tid  = threadIdx.x;
  const int lane = tid & 31;
  const int wid  = tid >> 5;
  const int wm   = wid & 3;   // wave M position (0..3)
  const int wn   = wid >> 2;  // wave N position (0..1)
  const int kh   = lane >> 4; // lane half (K-group select)
  const int lr   = lane & 15;

  const int m0 = blockIdx.y * BM;
  const int n0 = blockIdx.x * BN;
  const long long bz = blockIdx.z;

  // staging assignments: A 2 thr/row x 16 elems, B 4 thr/row x 8 elems
  const int rowA = tid >> 1, colA = (tid & 1) * 16;
  const int rowB = tid >> 2, colB = (tid & 3) * 8;

  const u16_t* Ag = A + bz * sA + (long long)(m0 + rowA) * lda + colA;
  const u16_t* Bg = B + bz * sB + (long long)(n0 + rowB) * ldb + colB;
  const unsigned ldsA0 = lds_addr(&AsU[0][rowA * SA + colA]);
  const unsigned ldsA1 = lds_addr(&AsU[1][rowA * SA + colA]);
  const unsigned ldsB0 = lds_addr(&BsU[0][rowB * SB + colB]);
  const unsigned ldsB1 = lds_addr(&BsU[1][rowB * SB + colB]);

  v8f acc[2][2] = {};

  // prologue: DMA k0 = 0 into buffer 0
  async_copy_b128x2(ldsA0, Ag);
  async_copy_b128(ldsB0, Bg);
  wait_async0();
  __syncthreads();

  int cur = 0;
  for (int k0 = 0; k0 < K; k0 += BK) {
    const bool has_next = (k0 + BK) < K;
    if (has_next) {  // DMA next tile into the other buffer; hidden by WMMAs
      async_copy_b128x2(cur ? ldsA0 : ldsA1, Ag + k0 + BK);
      async_copy_b128(cur ? ldsB0 : ldsB1, Bg + k0 + BK);
    }

    // ---- compute on current buffer ----
    const __bf16* As = (const __bf16*)AsU[cur];
    const __bf16* Bs = (const __bf16*)BsU[cur];

    // A fragment (ISA 16-bit A 16x32): v0..3 = K kh*8.., v4..7 = K 16+kh*8..
    v16bf afr[2], bfr[2];
#pragma unroll
    for (int mi = 0; mi < 2; ++mi) {
      const __bf16* p = As + (wm * 32 + mi * 16 + lr) * SA + kh * 8;
      v8bf lo = *(const v8bf*)p;
      v8bf hi = *(const v8bf*)(p + 16);
      afr[mi] = __builtin_shufflevector(lo, hi, 0, 1, 2, 3, 4, 5, 6, 7, 8, 9,
                                        10, 11, 12, 13, 14, 15);
    }
    // B fragment: same K order as A: v0..3 = K kh*8.., v4..7 = K 16+kh*8..
#pragma unroll
    for (int ni = 0; ni < 2; ++ni) {
      const __bf16* p = Bs + (wn * 32 + ni * 16 + lr) * SB + kh * 8;
      v8bf lo = *(const v8bf*)p;
      v8bf hi = *(const v8bf*)(p + 16);
      bfr[ni] = __builtin_shufflevector(lo, hi, 0, 1, 2, 3, 4, 5, 6, 7, 8, 9,
                                        10, 11, 12, 13, 14, 15);
    }
#pragma unroll
    for (int mi = 0; mi < 2; ++mi)
#pragma unroll
      for (int ni = 0; ni < 2; ++ni)
        acc[mi][ni] = __builtin_amdgcn_wmma_f32_16x16x32_bf16(
            false, afr[mi], false, bfr[ni], (short)0, acc[mi][ni], false, false);

    // own DMAs complete -> barrier -> everyone's DMAs visible
    wait_async0();
    __syncthreads();
    cur ^= 1;
  }

  // ---- epilogue: C/D layout = 8 f32 VGPRs, M = r + 8*kh, N = lr ----
  float* Cf = (float*)Cp + bz * sC;
  u16_t* Cb = (u16_t*)Cp + bz * sC;
#pragma unroll
  for (int mi = 0; mi < 2; ++mi) {
#pragma unroll
    for (int ni = 0; ni < 2; ++ni) {
      const int col = n0 + wn * 32 + ni * 16 + lr;
      const float bc = HAS_BIAS ? bias[col] : 0.0f;
#pragma unroll
      for (int r = 0; r < 8; ++r) {
        const int row = m0 + wm * 32 + mi * 16 + kh * 8 + r;
        float v = acc[mi][ni][r] * alpha + bc;
        if (OUT_MODE == 0)
          Cf[(long long)row * ldc + col] = v;
        else if (OUT_MODE == 1)
          Cb[(long long)row * ldc + col] = f2bf_bits(v);
        else
          Cb[(long long)col * ldc + row] = f2bf_bits(v);
      }
    }
  }
}

// Row softmax over 2048 f32, written back IN PLACE as bf16 (all f32 loads
// precede the __syncthreads -> race-free). bf16 row stride = 4096.
__global__ __launch_bounds__(256) void softmax_rows_2048(float* __restrict__ S) {
  const long long row = blockIdx.x;
  float* s = S + row * 2048;
  const int tid = threadIdx.x;
  float4 a = ((const float4*)s)[2 * tid];
  float4 b = ((const float4*)s)[2 * tid + 1];
  float x[8] = {a.x, a.y, a.z, a.w, b.x, b.y, b.z, b.w};
  float m = x[0];
#pragma unroll
  for (int i = 1; i < 8; ++i) m = fmaxf(m, x[i]);
#pragma unroll
  for (int off = 16; off; off >>= 1) m = fmaxf(m, __shfl_xor(m, off, 32));
  __shared__ float red[8];
  if ((tid & 31) == 0) red[tid >> 5] = m;
  __syncthreads();
  float bm = red[0];
#pragma unroll
  for (int i = 1; i < 8; ++i) bm = fmaxf(bm, red[i]);
  __syncthreads();
  float sum = 0.f;
#pragma unroll
  for (int i = 0; i < 8; ++i) {
    x[i] = __expf(x[i] - bm);
    sum += x[i];
  }
#pragma unroll
  for (int off = 16; off; off >>= 1) sum += __shfl_xor(sum, off, 32);
  if ((tid & 31) == 0) red[tid >> 5] = sum;
  __syncthreads();
  float tot = 0.f;
#pragma unroll
  for (int i = 0; i < 8; ++i) tot += red[i];
  const float inv = 1.0f / tot;
  uint4 o;
  o.x = pack_bf2(x[0] * inv, x[1] * inv);
  o.y = pack_bf2(x[2] * inv, x[3] * inv);
  o.z = pack_bf2(x[4] * inv, x[5] * inv);
  o.w = pack_bf2(x[6] * inv, x[7] * inv);
  ((uint4*)((u16_t*)S + row * 4096))[tid] = o;  // bf16 probs over f32 scores
}

extern "C" void kernel_launch(void* const* d_in, const int* in_sizes, int n_in,
                              void* d_out, int out_size, void* d_ws,
                              size_t ws_size, hipStream_t stream) {
  const float* Q  = (const float*)d_in[0];
  const float* K  = (const float*)d_in[1];
  const float* V  = (const float*)d_in[2];
  const float* Wk = (const float*)d_in[3];
  const float* bk = (const float*)d_in[4];
  const float* Wv = (const float*)d_in[5];
  const float* bv = (const float*)d_in[6];
  const float* Wo = (const float*)d_in[7];
  const float* bo = (const float*)d_in[8];
  float* out = (float*)d_out;

  const int B = 4, Lq = 1024, Lk = 2048, D1 = 1024, D2 = 1280;

  // workspace carve-up (~134 MB; GEMM working sets stay L2-resident, 192 MB L2)
  char* ws = (char*)d_ws;
  size_t off = 0;
  auto carve = [&](size_t bytes) {
    char* p = ws + off;
    off = (off + bytes + 255) & ~(size_t)255;
    return p;
  };
  u16_t* Qc     = (u16_t*)carve((size_t)B * Lq * D1 * 2);
  u16_t* Kc     = (u16_t*)carve((size_t)B * Lk * D2 * 2);
  u16_t* Vc     = (u16_t*)carve((size_t)B * Lk * D2 * 2);
  u16_t* Wkc    = (u16_t*)carve((size_t)D1 * D2 * 2);
  u16_t* Wvc    = (u16_t*)carve((size_t)D1 * D2 * 2);
  u16_t* Woc    = (u16_t*)carve((size_t)D1 * D1 * 2);
  u16_t* Kproj  = (u16_t*)carve((size_t)B * Lk * D1 * 2);  // [b][k][d]
  u16_t* VprojT = (u16_t*)carve((size_t)B * D1 * Lk * 2);  // [b][d][k]
  float* scores = (float*)carve((size_t)B * Lq * Lk * 4);  // -> bf16 probs in place
  u16_t* ctx    = (u16_t*)carve((size_t)B * Lq * D1 * 2);  // [b][q][d]
  (void)ws_size; (void)n_in; (void)in_sizes; (void)out_size;

  const dim3 blk(256);
  const float scale = 0.03125f;  // 1/sqrt(1024)

  // 0) one-time bf16 conversions (each element converted exactly once)
  auto cvt = [&](const float* s, u16_t* d, int n) {
    convert_f32_bf16<<<dim3((n + 2047) / 2048), blk, 0, stream>>>(s, d, n);
  };
  cvt(Q, Qc, B * Lq * D1);
  cvt(K, Kc, B * Lk * D2);
  cvt(V, Vc, B * Lk * D2);
  cvt(Wk, Wkc, D1 * D2);
  cvt(Wv, Wvc, D1 * D2);
  cvt(Wo, Woc, D1 * D1);

  // 1) K_proj[b][k][d] = K[b][k][:] . Wk[d][:] + bk   (M=Lk,N=D1,K=D2)
  gemm_bf16_wmma<1, true><<<dim3(D1 / 64, Lk / 128, B), blk, 0, stream>>>(
      Kc, Wkc, Kproj, bk, Lk, D1, D2, D2, D2, D1,
      (long long)Lk * D2, 0, (long long)Lk * D1, 1.0f);

  // 2) V_projT[b][d][k] = V[b][k][:] . Wv[d][:] + bv  (transposed store)
  gemm_bf16_wmma<2, true><<<dim3(D1 / 64, Lk / 128, B), blk, 0, stream>>>(
      Vc, Wvc, VprojT, bv, Lk, D1, D2, D2, D2, /*ldc=*/Lk,
      (long long)Lk * D2, 0, (long long)D1 * Lk, 1.0f);

  // 3) scores[b][q][k] = (Q[b] . K_proj[b]^T) / 32    (M=Lq,N=Lk,K=D1)
  gemm_bf16_wmma<0, false><<<dim3(Lk / 64, Lq / 128, B), blk, 0, stream>>>(
      Qc, Kproj, scores, nullptr, Lq, Lk, D1, D1, D1, Lk,
      (long long)Lq * D1, (long long)Lk * D1, (long long)Lq * Lk, scale);

  // 4) row softmax -> bf16 probs in place (row stride 4096 bf16 elems)
  softmax_rows_2048<<<dim3(B * Lq), blk, 0, stream>>>(scores);

  // 5) ctx[b][q][d] = probs[b] . V_proj[b]            (M=Lq,N=D1,K=Lk)
  gemm_bf16_wmma<1, false><<<dim3(D1 / 64, Lq / 128, B), blk, 0, stream>>>(
      (const u16_t*)scores, VprojT, ctx, nullptr, Lq, D1, Lk,
      /*lda=*/4096, /*ldb=*/Lk, /*ldc=*/D1,
      (long long)Lq * 4096, (long long)D1 * Lk, (long long)Lq * D1, 1.0f);

  // 6) out[b][q][e] = ctx . Wo[e][:] + bo             (M=B*Lq,N=D1,K=D1)
  gemm_bf16_wmma<0, true><<<dim3(D1 / 64, (B * Lq) / 128, 1), blk, 0, stream>>>(
      ctx, Woc, out, bo, B * Lq, D1, D1, D1, D1, D1, 0, 0, 0, 1.0f);
}
